// StateSpaceModel_39762807226711
// MI455X (gfx1250) — compile-verified, matrix operand order checked by reading, not run on hardware
//
#include <hip/hip_runtime.h>

#define D_MODEL 1024
#define CLIP_V  10.0f

typedef __attribute__((ext_vector_type(16))) __bf16         v16bf;
typedef __attribute__((ext_vector_type(8)))  float          v8f;
typedef __attribute__((ext_vector_type(4)))  unsigned short us4;
typedef __attribute__((ext_vector_type(4)))  unsigned int   u32x4;

__device__ __forceinline__ float clipf(float f) {
    return fminf(fmaxf(f, -CLIP_V), CLIP_V);
}

// fp32 -> bf16 round-to-nearest-even (finite inputs only)
__device__ __forceinline__ unsigned short f32_to_bf16(float f) {
    unsigned int u = __float_as_uint(f);
    u += 0x7FFFu + ((u >> 16) & 1u);
    return (unsigned short)(u >> 16);
}

union FragCast { u32x4 q[2]; us4 h[4]; v16bf v; };

__device__ __forceinline__ v8f wmma_bf16(v16bf a, v16bf b, v8f c) {
    return __builtin_amdgcn_wmma_f32_16x16x32_bf16(
        /*neg_a=*/false, a, /*neg_b=*/false, b,
        /*c_mod=*/(short)0, c, /*reuse_a=*/false, /*reuse_b=*/false);
}

// ---------------------------------------------------------------------------
// Prepass 1: clip(x) -> bf16, swizzled into A-fragment order.
// Fragment index: ((strip*32 + t)*32 + lane), each fragment = 32 bytes.
// strip = row/16; lane<16 holds M=strip*16+lane, K = t*32 + {0..7,16..23};
// lane>=16 same M rows, K = t*32 + {8..15,24..31}.
// ---------------------------------------------------------------------------
__global__ __launch_bounds__(256)
void prep_x_frag(const float* __restrict__ x, unsigned short* __restrict__ aws, int total) {
    const int flat = blockIdx.x * 256 + threadIdx.x;
    if (flat >= total) return;
    const int l = flat & 31;
    const int t = (flat >> 5) & 31;
    const int s = flat >> 10;
    const int m     = s * 16 + (l & 15);
    const int kbase = t * 32 + ((l >> 4) << 3);   // + khalf

    const float* src = x + (size_t)m * D_MODEL + kbase;
    FragCast fc;
    #pragma unroll
    for (int c = 0; c < 2; ++c) {                 // chunk 0: K+0..7, chunk 1: K+16..23
        const float4 f0 = *(const float4*)(src + c * 16);
        const float4 f1 = *(const float4*)(src + c * 16 + 4);
        us4 a, b;
        a[0] = f32_to_bf16(clipf(f0.x)); a[1] = f32_to_bf16(clipf(f0.y));
        a[2] = f32_to_bf16(clipf(f0.z)); a[3] = f32_to_bf16(clipf(f0.w));
        b[0] = f32_to_bf16(clipf(f1.x)); b[1] = f32_to_bf16(clipf(f1.y));
        b[2] = f32_to_bf16(clipf(f1.z)); b[3] = f32_to_bf16(clipf(f1.w));
        fc.h[2 * c + 0] = a;
        fc.h[2 * c + 1] = b;
    }
    u32x4* dst = (u32x4*)(aws + (size_t)flat * 16);
    dst[0] = fc.q[0];
    dst[1] = fc.q[1];
}

// ---------------------------------------------------------------------------
// Prepass 2: W/U/V -> bf16, swizzled into B-fragment order.
// Fragment index: (((g*64 + ntile)*32 + t)*32 + lane), 32 bytes each.
// ---------------------------------------------------------------------------
__global__ __launch_bounds__(256)
void prep_w_frag(const float* __restrict__ W, const float* __restrict__ U,
                 const float* __restrict__ V, unsigned short* __restrict__ bws) {
    const int flat = blockIdx.x * 256 + threadIdx.x;   // < 3*64*32*32 = 196608
    const int l = flat & 31;
    const int t = (flat >> 5) & 31;
    const int n = (flat >> 10) & 63;
    const int g = flat >> 16;
    const float* src = (g == 0) ? W : (g == 1) ? U : V;

    const int col   = n * 16 + (l & 15);
    const int kbase = t * 32 + ((l >> 4) << 3);

    FragCast fc;
    #pragma unroll
    for (int c = 0; c < 2; ++c) {
        #pragma unroll
        for (int p = 0; p < 2; ++p) {
            us4 h;
            #pragma unroll
            for (int i = 0; i < 4; ++i) {
                const int k = kbase + c * 16 + p * 4 + i;
                h[i] = f32_to_bf16(src[(size_t)k * D_MODEL + col]);
            }
            fc.h[2 * c + p] = h;
        }
    }
    u32x4* dst = (u32x4*)(bws + (size_t)flat * 16);
    dst[0] = fc.q[0];
    dst[1] = fc.q[1];
}

// ---------------------------------------------------------------------------
// Main GEMM: fragments loaded directly global -> VGPR (no LDS, no cvt).
// Block = 256 threads = 8 waves, tile 128(M) x 64(N).
// Wave = 2 M-strips x 2 N-tiles x 3 gates = 12 accumulators, 12 WMMAs/K-step.
// All 16 loads per K-step share TWO base pointers + constant IOFFSETs:
//   A: +mi*32768, +{0,16};  B: +g*2097152 + ni*32768 + {0,16};  t stride 1024.
// ---------------------------------------------------------------------------
__global__ __launch_bounds__(256)
void gemm_gated_frag(const unsigned short* __restrict__ aws,
                     const unsigned short* __restrict__ bws,
                     const float* __restrict__ x,
                     float* __restrict__ y)
{
    const int tid  = threadIdx.x;
    const int wave = tid >> 5;
    const int lane = tid & 31;
    const int lane15 = lane & 15;
    const bool hiHalf = (lane >= 16);

    const int waveM = wave >> 1;                      // 0..3
    const int waveN = wave & 1;                       // 0..1
    const int strip0 = blockIdx.y * 8 + waveM * 2;    // 2 strips (32 rows)
    const int nt0    = blockIdx.x * 4 + waveN * 2;    // 2 n-tiles (32 cols)

    // Base pointers for this wave's fragment streams (t = 0).
    const char* Ab = (const char*)aws + ((size_t)strip0 * 32 * 32 + lane) * 32;
    const char* Bb = (const char*)bws + ((size_t)nt0    * 32 * 32 + lane) * 32;

    v8f acc[2][2][3] = {};                            // [mi][ni][gate]

    #pragma unroll 1
    for (int t = 0; t < 32; ++t) {
        FragCast fa0, fa1;
        fa0.q[0] = *(const u32x4*)(Ab + 0);
        fa0.q[1] = *(const u32x4*)(Ab + 16);
        fa1.q[0] = *(const u32x4*)(Ab + 32768);
        fa1.q[1] = *(const u32x4*)(Ab + 32768 + 16);
        // prefetch next K-step's A fragment (global_prefetch_b8); no extra regs
        __builtin_prefetch(Ab + 1024, 0, 1);

        #pragma unroll
        for (int ni = 0; ni < 2; ++ni) {
            #pragma unroll
            for (int g = 0; g < 3; ++g) {
                const int off = g * 2097152 + ni * 32768;
                FragCast fb;
                fb.q[0] = *(const u32x4*)(Bb + off);
                fb.q[1] = *(const u32x4*)(Bb + off + 16);
                acc[0][ni][g] = wmma_bf16(fa0.v, fb.v, acc[0][ni][g]);
                acc[1][ni][g] = wmma_bf16(fa1.v, fb.v, acc[1][ni][g]);
            }
        }
        Ab += 1024;                                   // next K-step
        Bb += 1024;
    }

    // Epilogue: C/D layout: VGPR r, lane L -> M = r + 8*(L>=16), N = L&15.
    #pragma unroll
    for (int mi = 0; mi < 2; ++mi) {
        const int rowB = (strip0 + mi) * 16 + (hiHalf ? 8 : 0);
        #pragma unroll
        for (int ni = 0; ni < 2; ++ni) {
            const int col = (nt0 + ni) * 16 + lane15;
            #pragma unroll
            for (int r = 0; r < 8; ++r) {
                const int row = rowB + r;
                const float hA = tanhf(acc[mi][ni][0][r]);
                const float uA = 1.0f / (1.0f + __expf(-acc[mi][ni][1][r]));
                const float vA = 1.0f / (1.0f + __expf(-acc[mi][ni][2][r]));
                const float xc = clipf(x[(size_t)row * D_MODEL + col]);
                y[(size_t)row * D_MODEL + col] = clipf(uA * hA + vA * xc);
            }
        }
    }
}

// ---------------------------------------------------------------------------
// Fallback (fused, LDS-staged) if workspace is too small.
// ---------------------------------------------------------------------------
#define BM 128
#define BN 32
#define BK 32

__device__ __forceinline__ v16bf load_frag_lds(const unsigned short* rowBase, int khalf) {
    FragCast f;
    f.q[0] = *(const u32x4*)(rowBase + khalf);
    f.q[1] = *(const u32x4*)(rowBase + khalf + 16);
    return f.v;
}

__global__ __launch_bounds__(256, 2)
void gated_mlp_wmma_bf16(const float* __restrict__ x,
                         const float* __restrict__ W,
                         const float* __restrict__ U,
                         const float* __restrict__ V,
                         float* __restrict__ y)
{
    __shared__ unsigned short lA[BM * BK];
    __shared__ unsigned short lB[3][BN * BK];

    const int tid    = threadIdx.x;
    const int wave   = tid >> 5;
    const int lane   = tid & 31;
    const int lane15 = lane & 15;
    const bool hiHalf = (lane >= 16);
    const int khalf  = hiHalf ? 8 : 0;

    const int m0 = blockIdx.y * BM;
    const int n0 = blockIdx.x * BN;
    const int mRow = wave * 16;

    const float* Bsrc[3] = { W, U, V };
    v8f acc[2][3] = {};

    for (int k0 = 0; k0 < D_MODEL; k0 += BK) {
        __syncthreads();
        {
            const int kq   = (tid & 7) * 4;
            const int rsub = tid >> 3;
            #pragma unroll
            for (int r = 0; r < 4; ++r) {
                const int row = r * 32 + rsub;
                const float4 f = *(const float4*)(x + (size_t)(m0 + row) * D_MODEL + k0 + kq);
                us4 b;
                b[0] = f32_to_bf16(clipf(f.x));
                b[1] = f32_to_bf16(clipf(f.y));
                b[2] = f32_to_bf16(clipf(f.z));
                b[3] = f32_to_bf16(clipf(f.w));
                *(us4*)&lA[row * BK + kq] = b;
            }
        }
        {
            const int kk = tid >> 3;
            const int nq = (tid & 7) * 4;
            #pragma unroll
            for (int g = 0; g < 3; ++g) {
                const float4 f = *(const float4*)(Bsrc[g] + (size_t)(k0 + kk) * D_MODEL + n0 + nq);
                lB[g][(nq + 0) * BK + kk] = f32_to_bf16(f.x);
                lB[g][(nq + 1) * BK + kk] = f32_to_bf16(f.y);
                lB[g][(nq + 2) * BK + kk] = f32_to_bf16(f.z);
                lB[g][(nq + 3) * BK + kk] = f32_to_bf16(f.w);
            }
        }
        __syncthreads();

        const v16bf aF = load_frag_lds(&lA[(mRow + lane15) * BK], khalf);
        #pragma unroll
        for (int nt = 0; nt < 2; ++nt) {
            #pragma unroll
            for (int g = 0; g < 3; ++g) {
                const v16bf bF = load_frag_lds(&lB[g][(nt * 16 + lane15) * BK], khalf);
                acc[nt][g] = wmma_bf16(aF, bF, acc[nt][g]);
            }
        }
    }

    const int rowB = m0 + mRow + (hiHalf ? 8 : 0);
    #pragma unroll
    for (int nt = 0; nt < 2; ++nt) {
        const int col = n0 + nt * 16 + lane15;
        #pragma unroll
        for (int r = 0; r < 8; ++r) {
            const int row = rowB + r;
            const float hA = tanhf(acc[nt][0][r]);
            const float uA = 1.0f / (1.0f + __expf(-acc[nt][1][r]));
            const float vA = 1.0f / (1.0f + __expf(-acc[nt][2][r]));
            const float xc = clipf(x[(size_t)row * D_MODEL + col]);
            y[(size_t)row * D_MODEL + col] = clipf(uA * hA + vA * xc);
        }
    }
}

extern "C" void kernel_launch(void* const* d_in, const int* in_sizes, int n_in,
                              void* d_out, int out_size, void* d_ws, size_t ws_size,
                              hipStream_t stream) {
    const float* x = (const float*)d_in[0];
    const float* W = (const float*)d_in[1];
    const float* U = (const float*)d_in[2];
    const float* V = (const float*)d_in[3];
    float* y = (float*)d_out;

    const int rows   = in_sizes[0] / D_MODEL;            // 32768
    const int strips = rows / 16;                         // 2048
    const size_t aBytes = (size_t)strips * 32 * 32 * 32;  // 64 MiB
    const size_t bBytes = (size_t)3 * 64 * 32 * 32 * 32;  // 6 MiB

    if (ws_size >= aBytes + bBytes) {
        unsigned short* aws = (unsigned short*)d_ws;
        unsigned short* bws = (unsigned short*)((char*)d_ws + aBytes);

        const int totalA = strips * 1024;                 // one thread per 32B fragment
        prep_x_frag<<<(totalA + 255) / 256, dim3(256), 0, stream>>>(x, aws, totalA);
        prep_w_frag<<<(3 * 64 * 32 * 32) / 256, dim3(256), 0, stream>>>(W, U, V, bws);

        dim3 grid(D_MODEL / 64, rows / 128);              // (16, 256)
        gemm_gated_frag<<<grid, dim3(256), 0, stream>>>(aws, bws, x, y);
    } else {
        dim3 grid(D_MODEL / BN, rows / BM);               // (32, 256)
        gated_mlp_wmma_bf16<<<grid, dim3(256), 0, stream>>>(x, W, U, V, y);
    }
}
